// FISSA_68126771249636
// MI455X (gfx1250) — matrix operand order repaired and checked by reading, not compiled, not run on hardware
//
#include <hip/hip_runtime.h>
#include <hip/hip_bf16.h>
#include <math.h>

// ---------------------------------------------------------------------------
// FISSA fused forward for MI455X (gfx1250, wave32, WMMA).
// One workgroup (256 threads = 8 wave32) per batch element. All activations
// (50x64, padded to 64x64) live in LDS; every GEMM is a 4x4 grid of
// v_wmma_f32_16x16x32_f16 tiles with f32 accumulation.
// ---------------------------------------------------------------------------

typedef __attribute__((ext_vector_type(16))) _Float16 v16h;
typedef __attribute__((ext_vector_type(8)))  _Float16 v8h;
typedef __attribute__((ext_vector_type(8)))  float    v8f;

#define LSEQ     50
#define DDIM     64
#define STF      65          // f32 LDS row stride (floats)
#define STH      80          // f16 LDS row stride (halves) -> 160B rows, 32B aligned frags
#define NTHREADS 256
#define NWAVES   8
#define NEGF     (-4294967295.0f)   // -2^32 + 1
#define F16CLAMP 65504.0f

__device__ __forceinline__ float wave_sum(float v) {
#pragma unroll
  for (int o = 16; o; o >>= 1) v += __shfl_xor(v, o, 32);
  return v;
}
__device__ __forceinline__ float wave_max(float v) {
#pragma unroll
  for (int o = 16; o; o >>= 1) v = fmaxf(v, __shfl_xor(v, o, 32));
  return v;
}
__device__ __forceinline__ float sigm(float x) { return 1.0f / (1.0f + expf(-x)); }

// ---- staging helpers: f32 -> f16 LDS operand buffers -----------------------
// A operand: row-major [m][k] (stride STH).
__device__ __forceinline__ void stageA(const float* __restrict__ src, _Float16* __restrict__ Ah,
                                       int tid) {
  for (int i = tid; i < DDIM * DDIM; i += NTHREADS) {
    int r = i >> 6, c = i & 63;
    Ah[r * STH + c] = (_Float16)src[r * STF + c];
  }
}
__device__ __forceinline__ void stageA_clamp(const float* __restrict__ src, _Float16* __restrict__ Ah,
                                             int tid) {
  for (int i = tid; i < DDIM * DDIM; i += NTHREADS) {
    int r = i >> 6, c = i & 63;
    float v = src[r * STF + c];
    v = fminf(fmaxf(v, -F16CLAMP), F16CLAMP);
    Ah[r * STH + c] = (_Float16)v;
  }
}
// B operand stored column-major: Bh[n][k] = B[k][n].
// From a global row-major weight W[k][n] (64x64):
__device__ __forceinline__ void stageB_fromW(const float* __restrict__ Wg, _Float16* __restrict__ Bh,
                                             int tid) {
  for (int i = tid; i < DDIM * DDIM; i += NTHREADS) {
    int k = i >> 6, n = i & 63;                  // coalesced over n
    Bh[n * STH + k] = (_Float16)Wg[i];
  }
}
// From an LDS f32 buffer src[k][n] (transpose into Bh):
__device__ __forceinline__ void stageB_T(const float* __restrict__ src, _Float16* __restrict__ Bh,
                                         int tid) {
  for (int i = tid; i < DDIM * DDIM; i += NTHREADS) {
    int k = i >> 6, n = i & 63;
    Bh[n * STH + k] = (_Float16)src[k * STF + n];
  }
}
// From an LDS f32 buffer already holding B^T, i.e. src[n][k] (straight copy):
__device__ __forceinline__ void stageB_N(const float* __restrict__ src, _Float16* __restrict__ Bh,
                                         int tid) {
  for (int i = tid; i < DDIM * DDIM; i += NTHREADS) {
    int n = i >> 6, k = i & 63;
    Bh[n * STH + k] = (_Float16)src[n * STF + k];
  }
}

// ---- 64x64x64 GEMM via 16 WMMA tiles (each wave: 2 tiles, 2 K-steps) -------
__device__ __forceinline__ void gemm64(const _Float16* __restrict__ Ah,
                                       const _Float16* __restrict__ Bh,
                                       float* __restrict__ C, int tid) {
  const int wid = tid >> 5, lane = tid & 31;
  const int hl = lane >> 4, mn = lane & 15;
  for (int t = wid; t < 16; t += NWAVES) {
    const int tm = t >> 2, tn = t & 3;
    v8f acc = {0.f, 0.f, 0.f, 0.f, 0.f, 0.f, 0.f, 0.f};
    const _Float16* arow = Ah + (tm * 16 + mn) * STH;
    const _Float16* brow = Bh + (tn * 16 + mn) * STH;
#pragma unroll
    for (int kk = 0; kk < DDIM; kk += 32) {
      // A fragment (16x32 f16): lanes 0-15 rows K 0-7/16-23, lanes 16-31 K 8-15/24-31
      v8h a0 = *(const v8h*)(arow + kk + 8 * hl);
      v8h a1 = *(const v8h*)(arow + kk + 16 + 8 * hl);
      v16h a = __builtin_shufflevector(a0, a1, 0, 1, 2, 3, 4, 5, 6, 7,
                                       8, 9, 10, 11, 12, 13, 14, 15);
      // B fragment (32x16 f16), column-major staging: same K order as A
      v8h b0 = *(const v8h*)(brow + kk + 8 * hl);
      v8h b1 = *(const v8h*)(brow + kk + 16 + 8 * hl);
      v16h b = __builtin_shufflevector(b0, b1, 0, 1, 2, 3, 4, 5, 6, 7,
                                       8, 9, 10, 11, 12, 13, 14, 15);
      acc = __builtin_amdgcn_wmma_f32_16x16x32_f16(false, a, false, b,
                                                   (short)0, acc, false, false);
    }
#pragma unroll
    for (int v = 0; v < 8; v++)
      C[(tm * 16 + v + 8 * hl) * STF + tn * 16 + mn] = acc[v];
  }
}

// ---- per-row LayerNorm over D=64 (one wave per row) ------------------------
__device__ __forceinline__ void ln_rows(float* __restrict__ X, int tid, int nrows) {
  const int wid = tid >> 5, lane = tid & 31;
  for (int r = wid; r < nrows; r += NWAVES) {
    float a = X[r * STF + lane], b = X[r * STF + lane + 32];
    float mu = wave_sum(a + b) * (1.f / 64.f);
    float d0 = a - mu, d1 = b - mu;
    float var = wave_sum(d0 * d0 + d1 * d1) * (1.f / 64.f);
    float inv = rsqrtf(var + 1e-8f);
    X[r * STF + lane] = d0 * inv;
    X[r * STF + lane + 32] = d1 * inv;
  }
}

__global__ __launch_bounds__(NTHREADS) void fissa_fused(
    const int* __restrict__ user_items, const int* __restrict__ pos_items,
    const int* __restrict__ neg_items, const float* __restrict__ padding_mask,
    const float* __restrict__ item_emb, const float* __restrict__ pos_emb,
    const float* __restrict__ Qw, const float* __restrict__ Kw, const float* __restrict__ Vw,
    const float* __restrict__ c1w, const float* __restrict__ c1b,
    const float* __restrict__ c2w, const float* __restrict__ c2b,
    const float* __restrict__ query, const float* __restrict__ Klba,
    const float* __restrict__ Vlba,
    const float* __restrict__ l1w, const float* __restrict__ l1b,
    const float* __restrict__ l2w, const float* __restrict__ l2b,
    const float* __restrict__ gw, const float* __restrict__ gb,
    float* __restrict__ out, int Btot) {
  const int b = blockIdx.x;
  const int tid = threadIdx.x;
  const int wid = tid >> 5, lane = tid & 31;

  extern __shared__ __align__(16) unsigned char smem[];
  float* sIE  = (float*)smem;                 // 64 x STF  (gathered item embeddings)
  float* sXin = sIE + 64 * STF;               // post-LN inputs (kept for LBA)
  float* sXc  = sXin + 64 * STF;              // current activations x
  float* sB1  = sXc + 64 * STF;
  float* sB2  = sB1 + 64 * STF;
  float* sB3  = sB2 + 64 * STF;
  _Float16* sAh = (_Float16*)(sB3 + 64 * STF);   // 64 x STH halves
  _Float16* sBh = sAh + 64 * STH;                // 64 x STH halves (col-major B)
  float* sPM  = (float*)(sBh + 64 * STH);     // 64
  float* sQV  = sPM + 64;                     // 64 (query vector)
  float* sSC  = sQV + 64;                     // 64 (global-attn scores)
  float* sGLO = sSC + 64;                     // 64 (global vector)
  float* sGW  = sGLO + 64;                    // 192 (gating weights)

  // ---------- init: zero pad regions, load small tensors --------------------
  for (int i = tid; i < 64 * STF; i += NTHREADS) {
    sIE[i] = 0.f; sXin[i] = 0.f; sXc[i] = 0.f; sB1[i] = 0.f; sB2[i] = 0.f; sB3[i] = 0.f;
  }
  if (tid < 64) {
    sPM[tid] = (tid < LSEQ) ? padding_mask[b * LSEQ + tid] : 0.f;
    sQV[tid] = query[tid];
  }
  if (tid < 192) sGW[tid] = gw[tid];
  __syncthreads();

  // ---------- gather ie = item_embedding[user_items[b]] ----------------------
  for (int i = tid; i < LSEQ * DDIM; i += NTHREADS) {
    int r = i >> 6, c = i & 63;
    long row = (long)user_items[b * LSEQ + r];
    sIE[r * STF + c] = item_emb[row * DDIM + c];
  }
  __syncthreads();

  // ---------- inputs = LN((ie + pos_emb) * pm) -------------------------------
  for (int i = tid; i < LSEQ * DDIM; i += NTHREADS) {
    int r = i >> 6, c = i & 63;
    sXin[r * STF + c] = (sIE[r * STF + c] + pos_emb[i]) * sPM[r];
  }
  __syncthreads();
  ln_rows(sXin, tid, LSEQ);
  __syncthreads();
  for (int i = tid; i < 64 * STF; i += NTHREADS) sXc[i] = sXin[i];
  __syncthreads();

  // ---------- S=2 self-attention blocks --------------------------------------
  for (int s = 0; s < 2; s++) {
    const float* Qg = Qw + s * DDIM * DDIM;
    const float* Kg = Kw + s * DDIM * DDIM;
    const float* Vg = Vw + s * DDIM * DDIM;

    stageA(sXc, sAh, tid);
    stageB_fromW(Qg, sBh, tid);
    __syncthreads();
    gemm64(sAh, sBh, sB1, tid);               // Qx
    __syncthreads();
    stageB_fromW(Kg, sBh, tid);
    __syncthreads();
    gemm64(sAh, sBh, sB2, tid);               // Kx
    __syncthreads();
    stageB_fromW(Vg, sBh, tid);
    __syncthreads();
    gemm64(sAh, sBh, sB3, tid);               // Vx
    __syncthreads();

    // scores = Qx @ Kx^T : B operand column-major == Kx row-major
    stageA(sB1, sAh, tid);
    stageB_N(sB2, sBh, tid);
    __syncthreads();
    gemm64(sAh, sBh, sB1, tid);               // scores (overwrites Qx)
    __syncthreads();

    // scale, causal mask, key mask, * padding_mask
    for (int i = tid; i < DDIM * DDIM; i += NTHREADS) {
      int r = i >> 6, c = i & 63;
      float* p = &sB1[r * STF + c];
      if (r < LSEQ && c < LSEQ) {
        float v = *p * 0.125f;                // / sqrt(64)
        if (c > r) v = NEGF;                  // tril mask
        if (sPM[c] == 0.f) v = NEGF;          // key mask
        *p = v * sPM[r];
      } else {
        *p = 0.f;
      }
    }
    __syncthreads();

    // att = scores @ Vx + x ; LN
    stageA_clamp(sB1, sAh, tid);              // clamp NEG into f16 range
    stageB_T(sB3, sBh, tid);
    __syncthreads();
    gemm64(sAh, sBh, sB2, tid);               // att raw (overwrites Kx)
    __syncthreads();
    for (int i = tid; i < LSEQ * DDIM; i += NTHREADS) {
      int r = i >> 6, c = i & 63;
      sB2[r * STF + c] += sXc[r * STF + c];
    }
    __syncthreads();
    ln_rows(sB2, tid, LSEQ);
    __syncthreads();

    // h = relu(c1 @ att + c1_b)
    for (int i = tid; i < DDIM * DDIM; i += NTHREADS) {
      int r = i >> 6, c = i & 63;
      sAh[r * STH + c] =
          (_Float16)((r < LSEQ && c < LSEQ) ? c1w[s * LSEQ * LSEQ + r * LSEQ + c] : 0.f);
    }
    stageB_T(sB2, sBh, tid);
    __syncthreads();
    gemm64(sAh, sBh, sB1, tid);
    __syncthreads();
    for (int i = tid; i < LSEQ * DDIM; i += NTHREADS) {
      int r = i >> 6, c = i & 63;
      float v = sB1[r * STF + c] + c1b[s * LSEQ + r];
      sB1[r * STF + c] = v > 0.f ? v : 0.f;
    }
    __syncthreads();

    // ff = c2 @ h + c2_b + att ; * pm ; LN -> new x
    for (int i = tid; i < DDIM * DDIM; i += NTHREADS) {
      int r = i >> 6, c = i & 63;
      sAh[r * STH + c] =
          (_Float16)((r < LSEQ && c < LSEQ) ? c2w[s * LSEQ * LSEQ + r * LSEQ + c] : 0.f);
    }
    stageB_T(sB1, sBh, tid);
    __syncthreads();
    gemm64(sAh, sBh, sB3, tid);
    __syncthreads();
    for (int i = tid; i < LSEQ * DDIM; i += NTHREADS) {
      int r = i >> 6, c = i & 63;
      sXc[r * STF + c] = (sB3[r * STF + c] + c2b[s * LSEQ + r] + sB2[r * STF + c]) * sPM[r];
    }
    __syncthreads();
    ln_rows(sXc, tid, LSEQ);
    __syncthreads();
  }

  // ---------- global (LBA) attention -----------------------------------------
  stageA(sXin, sAh, tid);
  stageB_fromW(Klba, sBh, tid);
  __syncthreads();
  gemm64(sAh, sBh, sB1, tid);                 // Kx = inputs @ K_lba
  __syncthreads();
  stageB_fromW(Vlba, sBh, tid);
  __syncthreads();
  gemm64(sAh, sBh, sB2, tid);                 // Vx = inputs @ V_lba
  __syncthreads();

  if (tid < 64) {
    if (tid < LSEQ) {
      float acc = 0.f;
      for (int c = 0; c < DDIM; c++) acc += sQV[c] * sB1[tid * STF + c];
      sSC[tid] = (sPM[tid] == 0.f) ? NEGF : acc;
    } else {
      sSC[tid] = NEGF;
    }
  }
  __syncthreads();
  if (wid == 0) {                             // softmax over 50 (one wave)
    float s0 = (lane < LSEQ) ? sSC[lane] : -3.4e38f;
    float s1 = (lane + 32 < LSEQ) ? sSC[lane + 32] : -3.4e38f;
    float m = wave_max(fmaxf(s0, s1));
    float e0 = (lane < LSEQ) ? expf(s0 - m) : 0.f;
    float e1 = (lane + 32 < LSEQ) ? expf(s1 - m) : 0.f;
    float den = wave_sum(e0 + e1);
    if (lane < LSEQ) sSC[lane] = e0 / den;
    if (lane + 32 < LSEQ) sSC[lane + 32] = e1 / den;
  }
  __syncthreads();
  if (tid < DDIM) {                           // glo = sc @ Vx
    float acc = 0.f;
    for (int l = 0; l < LSEQ; l++) acc += sSC[l] * sB2[l * STF + tid];
    sGLO[tid] = acc;
  }
  __syncthreads();
  if (wid == 0) {                             // LN, scalar FFN, LN
    float a0 = sGLO[lane], a1 = sGLO[lane + 32];
    float mu = wave_sum(a0 + a1) * (1.f / 64.f);
    float d0 = a0 - mu, d1 = a1 - mu;
    float var = wave_sum(d0 * d0 + d1 * d1) * (1.f / 64.f);
    float inv = rsqrtf(var + 1e-8f);
    a0 = d0 * inv; a1 = d1 * inv;
    float w1 = l1w[0], bb1 = l1b[0], w2 = l2w[0], bb2 = l2b[0];
    float h0 = fmaxf(w1 * a0 + bb1, 0.f), h1 = fmaxf(w1 * a1 + bb1, 0.f);
    a0 = w2 * h0 + bb2 + a0; a1 = w2 * h1 + bb2 + a1;
    mu = wave_sum(a0 + a1) * (1.f / 64.f);
    d0 = a0 - mu; d1 = a1 - mu;
    var = wave_sum(d0 * d0 + d1 * d1) * (1.f / 64.f);
    inv = rsqrtf(var + 1e-8f);
    sGLO[lane] = d0 * inv; sGLO[lane + 32] = d1 * inv;
  }
  __syncthreads();

  // ---------- gating + output (one wave per (l, pos/neg) task) ---------------
  const float gbias = gb[0];
  for (int task = wid; task < 2 * LSEQ; task += NWAVES) {
    const int l = task >> 1, p = task & 1;
    const int* items = p ? neg_items : pos_items;
    long row = (long)items[b * LSEQ + l];
    float m0 = item_emb[row * DDIM + lane];
    float m1 = item_emb[row * DDIM + lane + 32];
    float ie0 = sIE[l * STF + lane], ie1 = sIE[l * STF + lane + 32];
    float g0 = sGLO[lane], g1 = sGLO[lane + 32];

    float part = ie0 * sGW[lane] + ie1 * sGW[lane + 32]
               + g0 * sGW[64 + lane] + g1 * sGW[96 + lane]
               + m0 * sGW[128 + lane] + m1 * sGW[160 + lane];
    float logit = wave_sum(part) + gbias;
    float g = sigm(sigm(logit));

    float o0 = (sXc[l * STF + lane] * g + g0 * (1.f - g)) * sPM[l];
    float o1 = (sXc[l * STF + lane + 32] * g + g1 * (1.f - g)) * sPM[l];
    float mu = wave_sum(o0 + o1) * (1.f / 64.f);
    float d0 = o0 - mu, d1 = o1 - mu;
    float var = wave_sum(d0 * d0 + d1 * d1) * (1.f / 64.f);
    float inv = rsqrtf(var + 1e-8f);
    float res = wave_sum(d0 * inv * m0 + d1 * inv * m1);
    if (lane == 0) out[(size_t)p * Btot * LSEQ + (size_t)b * LSEQ + l] = res;
  }
}

extern "C" void kernel_launch(void* const* d_in, const int* in_sizes, int n_in,
                              void* d_out, int out_size, void* d_ws, size_t ws_size,
                              hipStream_t stream) {
  const int B = in_sizes[0] / LSEQ;
  constexpr size_t SMEM_BYTES =
      (size_t)6 * 64 * STF * sizeof(float) +        // six f32 tile buffers
      (size_t)2 * 64 * STH * sizeof(_Float16) +     // A/B f16 operand buffers
      (size_t)(4 * 64 + 192) * sizeof(float);       // pm/qv/sc/glo/gw

  hipFuncSetAttribute(reinterpret_cast<const void*>(fissa_fused),
                      hipFuncAttributeMaxDynamicSharedMemorySize, (int)SMEM_BYTES);

  fissa_fused<<<dim3(B), dim3(NTHREADS), SMEM_BYTES, stream>>>(
      (const int*)d_in[0], (const int*)d_in[1], (const int*)d_in[2],
      (const float*)d_in[3], (const float*)d_in[4], (const float*)d_in[5],
      (const float*)d_in[6], (const float*)d_in[7], (const float*)d_in[8],
      (const float*)d_in[9], (const float*)d_in[10], (const float*)d_in[11],
      (const float*)d_in[12], (const float*)d_in[13], (const float*)d_in[14],
      (const float*)d_in[15], (const float*)d_in[16], (const float*)d_in[17],
      (const float*)d_in[18], (const float*)d_in[19], (const float*)d_in[20],
      (const float*)d_in[21], (float*)d_out, B);
}
